// S2VT_17033840296337
// MI455X (gfx1250) — matrix operand order repaired and checked by reading, not compiled, not run on hardware
//
#include <hip/hip_runtime.h>
#include <cstdint>
#include <cstddef>

// ---------------------------------------------------------------------------
// S2VT on MI455X (gfx1250): bf16 WMMA, fused gates-GEMM + LSTM cell.
//   B=64, TV=80, TC=40, H=1024, V=20000
// All fp32->bf16 conversion hoisted out of the 120-step serial recurrence.
// ---------------------------------------------------------------------------

typedef __attribute__((ext_vector_type(16))) __bf16 v16bf;
typedef __attribute__((ext_vector_type(8)))  float  v8f;

#define B_   64
#define TV_  80
#define TC_  40
#define H_   1024
#define V_   20000
#define G4_  4096   // 4*H (gate width)

__device__ __forceinline__ unsigned short f2bf(float x) {
  unsigned int u = __float_as_uint(x);
  u += 0x7fffu + ((u >> 16) & 1u);        // round-to-nearest-even
  return (unsigned short)(u >> 16);
}

__device__ __forceinline__ float sigm(float x) { return 1.0f / (1.0f + __expf(-x)); }

union FragBF {
  uint4 u[2];
  unsigned short h[16];
  v16bf v;
};

__device__ __forceinline__ v8f wmma_bf16(const FragBF& a, const FragBF& b, v8f c) {
  return __builtin_amdgcn_wmma_f32_16x16x32_bf16(false, a.v, false, b.v,
                                                 (short)0, c, false, false);
}

// A fragment (16x32 bf16, contiguous layout: lanes 0-15 -> K {0..15},
// lanes 16-31 -> K {16..31}) from a row-major bf16 row.
__device__ __forceinline__ FragBF load_a_bf16(const unsigned short* __restrict__ row,
                                              int kt, int g) {
  FragBF a;
  int k0 = kt * 32 + g * 16;
  a.u[0] = *(const uint4*)(row + k0);
  a.u[1] = *(const uint4*)(row + k0 + 8);
  return a;
}

// B fragment: 32 contiguous bytes per lane from the packed tile.
__device__ __forceinline__ FragBF load_b(const unsigned short* __restrict__ bp) {
  FragBF b;
  b.u[0] = *(const uint4*)(bp);
  b.u[1] = *(const uint4*)(bp + 8);
  return b;
}

// ---------------------------------------------------------------------------
// Pack fp32 weights [N, Ksrc] (two stacked sources along K) into bf16 WMMA-B
// fragment tiles: tile (nt,kt) = 32 lanes x 16 halves contiguous; per lane
// n = nt*16 + (lane&15); k = kt*32 + (lane>>4)*16 + j.  Tile order [nt][kt].
// ---------------------------------------------------------------------------
__global__ void pack_weights(const float* __restrict__ Wa, int Ka,
                             const float* __restrict__ Wb, int Kb,
                             unsigned short* __restrict__ dst,
                             int Ktiles, int total)
{
  int d = blockIdx.x * blockDim.x + threadIdx.x;
  if (d >= total) return;
  int j    = d & 15;
  int l    = (d >> 4) & 31;
  int tile = d >> 9;
  int kt   = tile % Ktiles;
  int nt   = tile / Ktiles;
  int n    = nt * 16 + (l & 15);
  int k    = kt * 32 + (l >> 4) * 16 + j;
  float v = (k < Ka) ? Wa[(size_t)n * Ka + k]
                     : Wb[(size_t)n * Kb + (k - Ka)];
  dst[d] = f2bf(v);
}

__global__ void add_bias(const float* __restrict__ a, const float* __restrict__ b,
                         float* __restrict__ out, int n)
{
  int i = blockIdx.x * blockDim.x + threadIdx.x;
  if (i < n) out[i] = a[i] + b[i];
}

// Bulk fp32 -> bf16 conversion (one-shot, bandwidth-bound).
__global__ void cvt_bf16(const float* __restrict__ src,
                         unsigned short* __restrict__ dst, int total)
{
  int i = blockIdx.x * blockDim.x + threadIdx.x;
  if (i < total) dst[i] = f2bf(src[i]);
}

// Pre-gather all decoder embeddings: emb[tc, b, k] = bf16(Wemb[captions[b,tc], k]).
__global__ void gather_emb(const float* __restrict__ Wemb,
                           const int*   __restrict__ captions,
                           unsigned short* __restrict__ emb)
{
  int i = blockIdx.x * blockDim.x + threadIdx.x;   // TC*B*H
  int k  = i & (H_ - 1);
  int bb = (i >> 10) & (B_ - 1);
  int tc = i >> 16;                                // i / (B_*H_)
  int tok = captions[bb * TC_ + tc];
  emb[i] = f2bf(Wemb[(size_t)tok * H_ + k]);
}

// ---------------------------------------------------------------------------
// Fused LSTM layer 1:  gates = [x_t | h1] @ W1^T + b  -> cell -> h1_out.
// Grid: 64 blocks (one 16-wide n-group each), 256 threads = 8 waves:
//   wave = (mt, ks): mt in 0..3 (M tile), ks in 0..1 (K split half).
// Each wave holds 4 accumulators (i,f,g,o chunk tiles for the same n-group);
// ks=1 partials reduce through LDS; ks=0 waves run the cell in registers.
// Decoder steps (xrow_base==null) skip the x phase entirely (x == 0).
// ---------------------------------------------------------------------------
__global__ void __launch_bounds__(256)
lstm_layer1(const unsigned short* __restrict__ xbf,  // bf16 action_feat [B,TV,H] or null
            int t,
            const unsigned short* __restrict__ hin,  // h1 (t-1) [64,1024] bf16
            const unsigned short* __restrict__ Wp,   // packed, Ktiles=64
            const float* __restrict__ bias,          // [4096]
            float* __restrict__ c,                   // c1 [64,1024] f32 (in-place)
            unsigned short* __restrict__ hout)       // h1 (t) [64,1024] bf16
{
  __shared__ float red[4][32][32];                   // [mt][elem][lane], conflict-free
  int lane = threadIdx.x & 31;
  int w    = threadIdx.x >> 5;
  int mt   = w & 3;
  int ks   = w >> 2;
  int ng   = blockIdx.x;
  int l15  = lane & 15;
  int g    = lane >> 4;
  int n    = ng * 16 + l15;
  int m    = mt * 16 + l15;                          // A-fragment row

  v8f acc[4];
#pragma unroll
  for (int cg = 0; cg < 4; ++cg) {
    float bv = (ks == 0) ? bias[cg * H_ + n] : 0.0f;
#pragma unroll
    for (int r = 0; r < 8; ++r) acc[cg][r] = bv;
  }

  const unsigned short* bpc[4];
#pragma unroll
  for (int cg = 0; cg < 4; ++cg)
    bpc[cg] = Wp + ((size_t)(cg * 64 + ng) * 64) * 512 + lane * 16;

  // Phase A (encoder only): x part, weight ktiles [0,32), split by ks.
  if (xbf) {
    const unsigned short* xrow = xbf + ((size_t)m * TV_ + t) * H_;
    for (int i = 0; i < 16; ++i) {
      int kt = ks * 16 + i;
      FragBF a = load_a_bf16(xrow, kt, g);
#pragma unroll
      for (int cg = 0; cg < 4; ++cg) {
        FragBF b = load_b(bpc[cg] + (size_t)kt * 512);
        acc[cg] = wmma_bf16(a, b, acc[cg]);
      }
    }
  }

  // Phase B: h part, weight ktiles [32,64), data ktiles [0,32), split by ks.
  {
    const unsigned short* hrow = hin + (size_t)m * H_;
    for (int i = 0; i < 16; ++i) {
      int dkt = ks * 16 + i;
      FragBF a = load_a_bf16(hrow, dkt, g);
#pragma unroll
      for (int cg = 0; cg < 4; ++cg) {
        FragBF b = load_b(bpc[cg] + (size_t)(32 + dkt) * 512);
        acc[cg] = wmma_bf16(a, b, acc[cg]);
      }
    }
  }

  // K-split reduction through LDS, then in-register LSTM cell (ks==0 waves).
  if (ks == 1) {
#pragma unroll
    for (int cg = 0; cg < 4; ++cg)
#pragma unroll
      for (int r = 0; r < 8; ++r) red[mt][cg * 8 + r][lane] = acc[cg][r];
  }
  __syncthreads();
  if (ks == 0) {
#pragma unroll
    for (int cg = 0; cg < 4; ++cg)
#pragma unroll
      for (int r = 0; r < 8; ++r) acc[cg][r] += red[mt][cg * 8 + r][lane];
#pragma unroll
    for (int r = 0; r < 8; ++r) {
      int mm = mt * 16 + g * 8 + r;                  // C-fragment row
      size_t ci = (size_t)mm * H_ + n;
      float gi = sigm(acc[0][r]);
      float gf = sigm(acc[1][r]);
      float gg = tanhf(acc[2][r]);
      float go = sigm(acc[3][r]);
      float cn = gf * c[ci] + gi * gg;
      c[ci] = cn;
      hout[ci] = f2bf(go * tanhf(cn));
    }
  }
}

// ---------------------------------------------------------------------------
// Fused LSTM layer 2:  gates = [word | h1_t | h2_{t-1}] @ W2^T + b -> cell.
// Word phase reads the pre-gathered bf16 embedding slab (null in encoder).
// ---------------------------------------------------------------------------
__global__ void __launch_bounds__(256)
lstm_layer2(const unsigned short* __restrict__ embrow, // emb slab row base [B,H] bf16 or null
            const unsigned short* __restrict__ h1,     // h1 (t) bf16
            const unsigned short* __restrict__ h2in,   // h2 (t-1) bf16
            const unsigned short* __restrict__ Wp,     // packed, Ktiles=96
            const float* __restrict__ bias,
            float* __restrict__ c,                     // c2 (in-place)
            unsigned short* __restrict__ hout,         // h2 (t)
            unsigned short* __restrict__ hist)         // H2all slot (decoder) or null
{
  __shared__ float red[4][32][32];
  int lane = threadIdx.x & 31;
  int w    = threadIdx.x >> 5;
  int mt   = w & 3;
  int ks   = w >> 2;
  int ng   = blockIdx.x;
  int l15  = lane & 15;
  int g    = lane >> 4;
  int n    = ng * 16 + l15;
  int m    = mt * 16 + l15;

  v8f acc[4];
#pragma unroll
  for (int cg = 0; cg < 4; ++cg) {
    float bv = (ks == 0) ? bias[cg * H_ + n] : 0.0f;
#pragma unroll
    for (int r = 0; r < 8; ++r) acc[cg][r] = bv;
  }

  const unsigned short* bpc[4];
#pragma unroll
  for (int cg = 0; cg < 4; ++cg)
    bpc[cg] = Wp + ((size_t)(cg * 64 + ng) * 96) * 512 + lane * 16;

  // Phase A (decoder only): word embedding, weight ktiles [0,32).
  if (embrow) {
    const unsigned short* erow = embrow + (size_t)m * H_;
    for (int i = 0; i < 16; ++i) {
      int kt = ks * 16 + i;
      FragBF a = load_a_bf16(erow, kt, g);
#pragma unroll
      for (int cg = 0; cg < 4; ++cg) {
        FragBF b = load_b(bpc[cg] + (size_t)kt * 512);
        acc[cg] = wmma_bf16(a, b, acc[cg]);
      }
    }
  }

  // Phase B: h1 part, weight ktiles [32,64).
  {
    const unsigned short* hrow = h1 + (size_t)m * H_;
    for (int i = 0; i < 16; ++i) {
      int dkt = ks * 16 + i;
      FragBF a = load_a_bf16(hrow, dkt, g);
#pragma unroll
      for (int cg = 0; cg < 4; ++cg) {
        FragBF b = load_b(bpc[cg] + (size_t)(32 + dkt) * 512);
        acc[cg] = wmma_bf16(a, b, acc[cg]);
      }
    }
  }

  // Phase C: h2 part, weight ktiles [64,96).
  {
    const unsigned short* hrow = h2in + (size_t)m * H_;
    for (int i = 0; i < 16; ++i) {
      int dkt = ks * 16 + i;
      FragBF a = load_a_bf16(hrow, dkt, g);
#pragma unroll
      for (int cg = 0; cg < 4; ++cg) {
        FragBF b = load_b(bpc[cg] + (size_t)(64 + dkt) * 512);
        acc[cg] = wmma_bf16(a, b, acc[cg]);
      }
    }
  }

  if (ks == 1) {
#pragma unroll
    for (int cg = 0; cg < 4; ++cg)
#pragma unroll
      for (int r = 0; r < 8; ++r) red[mt][cg * 8 + r][lane] = acc[cg][r];
  }
  __syncthreads();
  if (ks == 0) {
#pragma unroll
    for (int cg = 0; cg < 4; ++cg)
#pragma unroll
      for (int r = 0; r < 8; ++r) acc[cg][r] += red[mt][cg * 8 + r][lane];
#pragma unroll
    for (int r = 0; r < 8; ++r) {
      int mm = mt * 16 + g * 8 + r;
      size_t ci = (size_t)mm * H_ + n;
      float gi = sigm(acc[0][r]);
      float gf = sigm(acc[1][r]);
      float gg = tanhf(acc[2][r]);
      float go = sigm(acc[3][r]);
      float cn = gf * c[ci] + gi * gg;
      c[ci] = cn;
      unsigned short hb = f2bf(go * tanhf(cn));
      hout[ci] = hb;
      if (hist) hist[ci] = hb;
    }
  }
}

// ---------------------------------------------------------------------------
// Batched vocab projection: out[2560,20000] = H2all @ ew_W^T + ew_b.
// 2x2 register blocking per wave (2 M-tiles x 2 N-tiles, 4 WMMA per kt on
// 2 A + 2 B fragment loads) + prefetch of the L2-resident packed B stream.
// Grid (625, 20), 128 threads = 4 waves (each wave: M-tile pair).
// ---------------------------------------------------------------------------
__global__ void __launch_bounds__(128)
gemm_logits(const unsigned short* __restrict__ A,
            const unsigned short* __restrict__ Bp,
            const float* __restrict__ bias,
            float* __restrict__ Cout)
{
  const int K = H_, Ktiles = H_ / 32, N = V_;
  int lane = threadIdx.x & 31;
  int wave = threadIdx.x >> 5;
  int l15  = lane & 15;
  int g    = lane >> 4;
  int mt0  = (blockIdx.y * 4 + wave) * 2;   // 0..158
  int nt0  = blockIdx.x * 2;                // 0..1248
  int n0   = nt0 * 16 + l15;

  float b0 = bias[n0];
  float b1 = bias[n0 + 16];
  v8f a00, a01, a10, a11;
#pragma unroll
  for (int r = 0; r < 8; ++r) { a00[r] = b0; a01[r] = b1; a10[r] = b0; a11[r] = b1; }

  const unsigned short* Ar0 = A + (size_t)(mt0 * 16 + l15) * K;
  const unsigned short* Ar1 = Ar0 + (size_t)16 * K;
  const unsigned short* bp0 = Bp + ((size_t)nt0 * Ktiles) * 512 + lane * 16;
  const unsigned short* bp1 = bp0 + (size_t)Ktiles * 512;

  for (int kt = 0; kt < Ktiles; ++kt) {
    __builtin_prefetch(bp0 + 8 * 512, 0, 1);       // global_prefetch_b8
    __builtin_prefetch(bp1 + 8 * 512, 0, 1);
    FragBF x0 = load_a_bf16(Ar0, kt, g);
    FragBF x1 = load_a_bf16(Ar1, kt, g);
    FragBF w0 = load_b(bp0);
    FragBF w1 = load_b(bp1);
    bp0 += 512; bp1 += 512;
    a00 = wmma_bf16(x0, w0, a00);
    a01 = wmma_bf16(x0, w1, a01);
    a10 = wmma_bf16(x1, w0, a10);
    a11 = wmma_bf16(x1, w1, a11);
  }

  float* c0 = Cout + (size_t)(mt0 * 16 + g * 8) * N + n0;
  float* c1 = c0 + (size_t)16 * N;
#pragma unroll
  for (int r = 0; r < 8; ++r) {
    c0[(size_t)r * N]      = a00[r];
    c0[(size_t)r * N + 16] = a01[r];
    c1[(size_t)r * N]      = a10[r];
    c1[(size_t)r * N + 16] = a11[r];
  }
}

// ---------------------------------------------------------------------------
// Host orchestration (graph-capture safe: kernels + hipMemsetAsync only).
// ---------------------------------------------------------------------------
extern "C" void kernel_launch(void* const* d_in, const int* in_sizes, int n_in,
                              void* d_out, int out_size, void* d_ws, size_t ws_size,
                              hipStream_t stream)
{
  const float* action_feat = (const float*)d_in[1];
  const int*   captions    = (const int*)  d_in[3];
  const float* Wemb        = (const float*)d_in[5];
  const float* l1_Wih      = (const float*)d_in[6];
  const float* l1_Whh      = (const float*)d_in[7];
  const float* l1_bih      = (const float*)d_in[8];
  const float* l1_bhh      = (const float*)d_in[9];
  const float* l2_Wih      = (const float*)d_in[10];
  const float* l2_Whh      = (const float*)d_in[11];
  const float* l2_bih      = (const float*)d_in[12];
  const float* l2_bhh      = (const float*)d_in[13];
  const float* ew_W        = (const float*)d_in[14];
  const float* ew_b        = (const float*)d_in[15];

  char* ws = (char*)d_ws;
  size_t off = 0;
  auto take = [&](size_t bytes) -> char* {
    off = (off + 255) & ~(size_t)255;
    char* p = ws + off;
    off += bytes;
    return p;
  };

  unsigned short* W1p = (unsigned short*)take((size_t)2048 * G4_ * 2);    // 16.8 MB
  unsigned short* W2p = (unsigned short*)take((size_t)3072 * G4_ * 2);    // 25.2 MB
  unsigned short* WOp = (unsigned short*)take((size_t)H_ * V_ * 2);       // 41.0 MB
  float*          b1  = (float*)take((size_t)G4_ * 4);
  float*          b2  = (float*)take((size_t)G4_ * 4);
  unsigned short* xbf = (unsigned short*)take((size_t)B_ * TV_ * H_ * 2); // 10.5 MB
  unsigned short* emb = (unsigned short*)take((size_t)TC_ * B_ * H_ * 2); // 5.2 MB
  unsigned short* h1a = (unsigned short*)take((size_t)B_ * H_ * 2);
  unsigned short* h1b = (unsigned short*)take((size_t)B_ * H_ * 2);
  unsigned short* h2a = (unsigned short*)take((size_t)B_ * H_ * 2);
  unsigned short* h2b = (unsigned short*)take((size_t)B_ * H_ * 2);
  float*          c1  = (float*)take((size_t)B_ * H_ * 4);
  float*          c2  = (float*)take((size_t)B_ * H_ * 4);
  unsigned short* H2  = (unsigned short*)take((size_t)TC_ * B_ * H_ * 2); // decoder h2 history

  // ---- one-shot prep: weight packing, bias folding, input conversion ----
  {
    int total = 2048 * G4_;
    pack_weights<<<(total + 255) / 256, 256, 0, stream>>>(l1_Wih, 1024, l1_Whh, 1024, W1p, 64, total);
  }
  {
    int total = 3072 * G4_;
    pack_weights<<<(total + 255) / 256, 256, 0, stream>>>(l2_Wih, 2048, l2_Whh, 1024, W2p, 96, total);
  }
  {
    int total = H_ * V_;
    pack_weights<<<(total + 255) / 256, 256, 0, stream>>>(ew_W, 1024, nullptr, 1, WOp, 32, total);
  }
  add_bias<<<(G4_ + 255) / 256, 256, 0, stream>>>(l1_bih, l1_bhh, b1, G4_);
  add_bias<<<(G4_ + 255) / 256, 256, 0, stream>>>(l2_bih, l2_bhh, b2, G4_);
  {
    int total = B_ * TV_ * H_;
    cvt_bf16<<<(total + 255) / 256, 256, 0, stream>>>(action_feat, xbf, total);
  }
  {
    int total = TC_ * B_ * H_;
    gather_emb<<<(total + 255) / 256, 256, 0, stream>>>(Wemb, captions, emb);
  }

  // ---- zero recurrent state ----
  hipMemsetAsync(h1a, 0, (size_t)B_ * H_ * 2, stream);
  hipMemsetAsync(h1b, 0, (size_t)B_ * H_ * 2, stream);
  hipMemsetAsync(h2a, 0, (size_t)B_ * H_ * 2, stream);
  hipMemsetAsync(h2b, 0, (size_t)B_ * H_ * 2, stream);
  hipMemsetAsync(c1, 0, (size_t)B_ * H_ * 4, stream);
  hipMemsetAsync(c2, 0, (size_t)B_ * H_ * 4, stream);

  unsigned short* h1buf[2] = { h1a, h1b };
  unsigned short* h2buf[2] = { h2a, h2b };

  // ---- 120 sequential timesteps: 2 fused kernels each ----
  for (int t = 0; t < TV_ + TC_; ++t) {
    int cur = t & 1, nxt = cur ^ 1;
    int dec = (t >= TV_);
    const unsigned short* xsrc = dec ? nullptr : xbf;
    lstm_layer1<<<H_ / 16, 256, 0, stream>>>(xsrc, t, h1buf[cur], W1p, b1, c1, h1buf[nxt]);

    const unsigned short* erow = dec ? (emb + (size_t)(t - TV_) * B_ * H_) : nullptr;
    unsigned short* hist = dec ? (H2 + (size_t)(t - TV_) * B_ * H_) : nullptr;
    lstm_layer2<<<H_ / 16, 256, 0, stream>>>(erow, h1buf[nxt], h2buf[cur], W2p, b2, c2,
                                             h2buf[nxt], hist);
  }

  // ---- batched vocab projection ----
  gemm_logits<<<dim3(V_ / 32, (TC_ * B_) / 128), 128, 0, stream>>>(H2, WOp, ew_b, (float*)d_out);
}
